// E3Relax_40192303956686
// MI455X (gfx1250) — compile-verified, matrix operand order checked by reading, not run on hardware
//
#include <hip/hip_runtime.h>

// ---------------- problem constants (from reference) ----------------
#define Hh     256
#define H3     768
#define H2     512
#define Hhalf  128
#define NRBF   128
#define Nn     10000
#define NP     10016              /* Nn padded to multiple of 32 for BM=32 GEMM */
#define Ee     80000
#define Bb     8
#define Ll     3

#define SCALE_SILU  (1.0f/0.6f)
#define INV_SQRT_3  0.57735026918962576f
#define INV_SQRT_H  0.0625f               /* 1/sqrt(256) */
#define INV_SQRT_2  0.70710678118654752f
#define CUTOFF_INV  (1.0f/6.0f)
#define RBF_COEFF  (-8064.5f)             /* -0.5 * 127^2 */

// ---------------- vector types for WMMA ----------------
typedef __attribute__((ext_vector_type(16))) __bf16 v16bf;
typedef __attribute__((ext_vector_type(8)))  __bf16 bf16x8;
typedef __attribute__((ext_vector_type(8)))  float  v8f;
typedef __attribute__((ext_vector_type(4)))  float  f32x4;

__device__ __forceinline__ float fast_rcp(float x) { return __builtin_amdgcn_rcpf(x); }

__device__ __forceinline__ float scaled_silu(float t) {
  // t * sigmoid(t) / 0.6 using v_exp_f32 + v_rcp_f32 (avoid IEEE divide sequence)
  return t * fast_rcp(1.0f + __expf(-t)) * SCALE_SILU;
}

__device__ __forceinline__ float fast_tanh(float x) {
  // tanh(x) = 1 - 2/(exp(2x)+1)
  return 1.0f - 2.0f * fast_rcp(__expf(2.0f * x) + 1.0f);
}

// A fragment (16x32 bf16) from row-major f32: p = &A[row*K + kk + 8*h]
// elements 0..7 <- p[0..7], elements 8..15 <- p[16..23]  (per ISA layout)
__device__ __forceinline__ v16bf load_a_f32(const float* __restrict__ p) {
  f32x4 f0 = *(const f32x4*)(p);
  f32x4 f1 = *(const f32x4*)(p + 4);
  f32x4 f2 = *(const f32x4*)(p + 16);
  f32x4 f3 = *(const f32x4*)(p + 20);
  v16bf a;
#pragma unroll
  for (int t = 0; t < 4; ++t) {
    a[t]      = (__bf16)f0[t];
    a[4 + t]  = (__bf16)f1[t];
    a[8 + t]  = (__bf16)f2[t];
    a[12 + t] = (__bf16)f3[t];
  }
  return a;
}

// B fragment (32x16 bf16) from pre-transposed bf16 WT[Nc,K]: p = &WT[n*K + kk + 8*h]
__device__ __forceinline__ v16bf load_b_bf16(const __bf16* __restrict__ p) {
  bf16x8 lo = *(const bf16x8*)(p);
  bf16x8 hi = *(const bf16x8*)(p + 16);
  v16bf b;
#pragma unroll
  for (int t = 0; t < 8; ++t) { b[t] = lo[t]; b[8 + t] = hi[t]; }
  return b;
}

#define WMMA_BF16(a, b, c) \
  __builtin_amdgcn_wmma_f32_16x16x32_bf16(false, (a), false, (b), (short)0, (c), false, false)

// ---------------- generic GEMM: C[M,Nc] = act(A[M,K] @ W[K,Nc] + bias) ----------------
// A f32 row-major, WT bf16 [Nc,K] row-major (pre-transposed), C f32 row-major.
// Block = 256 threads (8 waves); each wave owns a 32x16 output (2 row-tiles sharing
// its B fragment each K-step); grid = (Nc/128, Mpad/32). Rows padded, no guards.
__global__ __launch_bounds__(256)
void gemm_kernel(const float* __restrict__ A, const __bf16* __restrict__ WT,
                 const float* __restrict__ bias, float* __restrict__ C,
                 int K, int Nc, int act) {
  const int wave = threadIdx.x >> 5;
  const int lane = threadIdx.x & 31;
  const int h    = lane >> 4;
  const int n0   = lane & 15;
  const int m0   = lane & 15;
  const size_t rowBase = (size_t)blockIdx.y * 32;
  const int    col     = blockIdx.x * 128 + wave * 16 + n0;

  v8f acc0 = {}, acc1 = {};
  const float*  arow0 = A + (rowBase + m0) * (size_t)K;
  const float*  arow1 = arow0 + (size_t)16 * K;
  const __bf16* brow  = WT + (size_t)col * K;
  for (int kk = 0; kk < K; kk += 32) {
    v16bf b  = load_b_bf16(brow + kk + 8 * h);
    v16bf a0 = load_a_f32(arow0 + kk + 8 * h);
    acc0 = WMMA_BF16(a0, b, acc0);
    v16bf a1 = load_a_f32(arow1 + kk + 8 * h);
    acc1 = WMMA_BF16(a1, b, acc1);
  }
  float bv = bias ? bias[col] : 0.0f;
#pragma unroll
  for (int r = 0; r < 8; ++r) {
    size_t m = rowBase + r + 8 * h;
    float v0 = acc0[r] + bv;
    float v1 = acc1[r] + bv;
    if (act == 1) { v0 = scaled_silu(v0); v1 = scaled_silu(v1); }
    C[m * (size_t)Nc + col]        = v0;
    C[(m + 16) * (size_t)Nc + col] = v1;
  }
}

// ---------------- weight convert + transpose: dst[n*K+k] = (bf16)src[k*Nc+n] ----------------
__global__ void wconvT_kernel(const float* __restrict__ src, __bf16* __restrict__ dst,
                              int K, int Nc) {
  int idx = blockIdx.x * 256 + threadIdx.x;
  if (idx >= K * Nc) return;
  int k = idx / Nc, n = idx - k * Nc;
  dst[(size_t)n * K + k] = (__bf16)src[idx];
}

// ---------------- shift[e] = cell_offsets[e] @ cell[edge_batch[e]] ----------------
__global__ void shift_kernel(const float* __restrict__ cell, const float* __restrict__ co,
                             const int* __restrict__ neigh, float* __restrict__ shift) {
  int e = blockIdx.x * 256 + threadIdx.x;
  if (e >= Ee) return;
  int b = Bb - 1, acc = 0;
  for (int q = 0; q < Bb; ++q) { acc += neigh[q]; if (e < acc) { b = q; break; } }
  float o0 = co[e * 3], o1 = co[e * 3 + 1], o2 = co[e * 3 + 2];
  const float* c = cell + (size_t)b * 9;
#pragma unroll
  for (int v = 0; v < 3; ++v)
    shift[e * 3 + v] = o0 * c[v] + o1 * c[3 + v] + o2 * c[6 + v];
}

// ---------------- cat[n] = [atom_emb[z[n]], tags_emb[tags[n]]] ----------------
__global__ void embed_cat_kernel(const float* __restrict__ ae, const float* __restrict__ te,
                                 const int* __restrict__ z, const int* __restrict__ tg,
                                 float* __restrict__ cat) {
  int idx = blockIdx.x * 256 + threadIdx.x;
  if (idx >= Nn * H2) return;
  int n = idx >> 9, k = idx & 511;
  cat[idx] = (k < Hh) ? ae[(size_t)z[n] * Hh + k] : te[(size_t)tg[n] * Hh + (k - Hh)];
}

// ---------------- fused edge kernel ----------------
// Per block: 16 edges, 8 waves. Wave w owns col-tiles {w, w+8, ..., w+40} so that the
// x1/x2/x3 tiles (t, t+16, t+32) of each channel group live in the SAME wave's
// accumulators -> fully register-resident epilogue, no [16,768] LDS staging.
// LDS: only the padded bf16 RBF tile (~5KB) -> high WGP occupancy.
__global__ __launch_bounds__(256)
void edge_kernel(const float* __restrict__ pos, const float* __restrict__ shift,
                 const int* __restrict__ eidx, const float* __restrict__ x_h,
                 const float* __restrict__ vec, const __bf16* __restrict__ epT,
                 const float* __restrict__ eb, float* __restrict__ d_vec,
                 float* __restrict__ d_x) {
  __shared__ __align__(16) __bf16 sA[16][136];   // 272B row => conflict-free b128 reads
  __shared__ float sDs[16];
  __shared__ float sEv[16][3];
  __shared__ int   sJ[16], sI[16];

  const int tid = threadIdx.x;
  const int e0  = blockIdx.x * 16;

  if (tid < 16) {
    int e = e0 + tid;
    int j = eidx[e], i = eidx[Ee + e];
    float dx = pos[j * 3 + 0] + shift[e * 3 + 0] - pos[i * 3 + 0];
    float dy = pos[j * 3 + 1] + shift[e * 3 + 1] - pos[i * 3 + 1];
    float dz = pos[j * 3 + 2] + shift[e * 3 + 2] - pos[i * 3 + 2];
    float d  = sqrtf(dx * dx + dy * dy + dz * dz);
    float inv = 1.0f / d;
    sDs[tid] = d * CUTOFF_INV;
    sEv[tid][0] = -dx * inv; sEv[tid][1] = -dy * inv; sEv[tid][2] = -dz * inv;
    sJ[tid] = j; sI[tid] = i;
  }
  __syncthreads();

  { // RBF: 16 threads per edge, 8 basis values each
    int m  = tid >> 4;
    int r0 = (tid & 15) * 8;
    float ds = sDs[m];
    float env = 0.0f;
    if (ds < 1.0f) {
      float d2 = ds * ds;
      float d5 = d2 * d2 * ds;
      env = 1.0f - 21.0f * d5 + 35.0f * d5 * ds - 15.0f * d5 * d2;
    }
#pragma unroll
    for (int q = 0; q < 8; ++q) {
      float off = (float)(r0 + q) * (1.0f / 127.0f);
      float t = ds - off;
      sA[m][r0 + q] = (__bf16)(env * __expf(RBF_COEFF * t * t));
    }
  }
  __syncthreads();

  const int wave = tid >> 5, lane = tid & 31;
  const int h = lane >> 4, n0 = lane & 15, m0 = lane & 15;

  // 6 accumulators: tile jx -> col tile (wave + 8*jx)
  v8f acc[6] = {};
#pragma unroll
  for (int kt = 0; kt < 4; ++kt) {
    const int kk = kt * 32;
    const __bf16* pa = &sA[m0][kk + 8 * h];
    bf16x8 lo = *(const bf16x8*)pa;
    bf16x8 hi = *(const bf16x8*)(pa + 16);
    v16bf a;
#pragma unroll
    for (int t = 0; t < 8; ++t) { a[t] = lo[t]; a[8 + t] = hi[t]; }
#pragma unroll
    for (int jx = 0; jx < 6; ++jx) {
      const int col = (wave + 8 * jx) * 16 + n0;
      v16bf b = load_b_bf16(epT + (size_t)col * NRBF + kk + 8 * h);
      acc[jx] = WMMA_BF16(a, b, acc[jx]);
    }
  }

  // register epilogue: channel group g in {0,1}: c = 16*(wave+8g)+n0 in [0,256)
  // x1 = acc[g], x2 = acc[2+g], x3 = acc[4+g]
#pragma unroll
  for (int g = 0; g < 2; ++g) {
    const int c = (wave + 8 * g) * 16 + n0;
    const float eb1 = eb[c];
    const float eb2 = eb[Hh + c];
    const float eb3 = eb[2 * Hh + c];
#pragma unroll
    for (int r = 0; r < 8; ++r) {
      const int m = r + 8 * h;
      const int j = sJ[m], i = sI[m];
      const size_t jb = (size_t)j * H3, ib = (size_t)i * H3;
      float x1 = x_h[jb + c]          * (acc[g][r]     + eb1) * INV_SQRT_3;
      float x2 = x_h[jb + Hh + c]     * (acc[2 + g][r] + eb2) * INV_SQRT_3;
      float x3 = x_h[jb + 2 * Hh + c] * (acc[4 + g][r] + eb3) * INV_SQRT_3;
      atomicAdd(&d_x[(size_t)i * Hh + c], x3);
#pragma unroll
      for (int v = 0; v < 3; ++v) {
        float vj = vec[jb + v * Hh + c];
        atomicAdd(&d_vec[ib + v * Hh + c], (x1 * vj + x2 * sEv[m][v]) * INV_SQRT_H);
      }
    }
  }
}

// ---------------- x = (x + d_x)/sqrt2 ; vec += d_vec ----------------
__global__ void combine_kernel(float* __restrict__ x, const float* __restrict__ dx,
                               float* __restrict__ vec, const float* __restrict__ dv) {
  size_t idx = (size_t)blockIdx.x * 256 + threadIdx.x;
  if (idx >= (size_t)Nn * H3) return;
  vec[idx] += dv[idx];
  if (idx < (size_t)Nn * Hh) x[idx] = (x[idx] + dx[idx]) * INV_SQRT_2;
}

// ---------------- hcat = [x, ||vec2||] ----------------
__global__ void vnorm_cat_kernel(const float* __restrict__ x, const float* __restrict__ v12,
                                 float* __restrict__ hcat) {
  size_t idx = (size_t)blockIdx.x * 256 + threadIdx.x;
  if (idx >= (size_t)Nn * Hh) return;
  size_t n = idx >> 8; int c = (int)(idx & 255);
  float s = 1e-8f;
#pragma unroll
  for (int v = 0; v < 3; ++v) {
    float t = v12[(n * 3 + v) * H2 + Hh + c];
    s += t * t;
  }
  hcat[n * H2 + c]      = x[idx];
  hcat[n * H2 + Hh + c] = sqrtf(s);
}

// ---------------- self-update ----------------
__global__ void selfupd_kernel(float* __restrict__ x, float* __restrict__ vec,
                               const float* __restrict__ hbuf, const float* __restrict__ v12) {
  size_t idx = (size_t)blockIdx.x * 256 + threadIdx.x;
  if (idx >= (size_t)Nn * Hh) return;
  size_t n = idx >> 8; int c = (int)(idx & 255);
  float xv1 = hbuf[n * H3 + c];
  float xv2 = hbuf[n * H3 + Hh + c];
  float xv3 = hbuf[n * H3 + 2 * Hh + c];
  float xo  = x[idx];
  float dx  = xv2 * INV_SQRT_2 + xo * fast_tanh(xv3);
  x[idx] = (xo + dx) * INV_SQRT_2;
#pragma unroll
  for (int v = 0; v < 3; ++v)
    vec[(n * 3 + v) * (size_t)Hh + c] += xv1 * v12[(n * 3 + v) * (size_t)H2 + c];
}

// ---------------- structure update: softmax + pos delta ----------------
__global__ __launch_bounds__(256)
void struct_kernel(const float* __restrict__ logits, const float* __restrict__ vec,
                   const float* __restrict__ pdw, float* __restrict__ pos_cur,
                   float* __restrict__ out) {
  __shared__ float red[256];
  const int n = blockIdx.x, t = threadIdx.x;
  float lg = logits[(size_t)n * Hh + t];
  red[t] = lg; __syncthreads();
  for (int s = 128; s > 0; s >>= 1) { if (t < s) red[t] = fmaxf(red[t], red[t + s]); __syncthreads(); }
  float mx = red[0]; __syncthreads();
  float ex = __expf(lg - mx);
  red[t] = ex; __syncthreads();
  for (int s = 128; s > 0; s >>= 1) { if (t < s) red[t] += red[t + s]; __syncthreads(); }
  float a = ex * fast_rcp(red[0]); __syncthreads();
  float w = pdw[t] + a;
  for (int v = 0; v < 3; ++v) {
    red[t] = vec[(size_t)n * H3 + v * Hh + t] * w; __syncthreads();
    for (int s = 128; s > 0; s >>= 1) { if (t < s) red[t] += red[t + s]; __syncthreads(); }
    if (t == 0) {
      float p = pos_cur[n * 3 + v] + red[0] * 0.5f;
      pos_cur[n * 3 + v] = p;
      out[n * 3 + v] = p;
    }
    __syncthreads();
  }
}

// ---------------- host launcher ----------------
extern "C" void kernel_launch(void* const* d_in, const int* in_sizes, int n_in,
                              void* d_out, int out_size, void* d_ws, size_t ws_size,
                              hipStream_t stream) {
  (void)in_sizes; (void)n_in; (void)out_size; (void)ws_size;
  const float* pos      = (const float*)d_in[0];
  const float* cell     = (const float*)d_in[1];
  const float* cofs     = (const float*)d_in[2];
  const float* atom_emb = (const float*)d_in[3];
  const float* tags_emb = (const float*)d_in[4];
  const float* comb_w   = (const float*)d_in[5];
  const float* comb_b   = (const float*)d_in[6];
  const float* xp_w1    = (const float*)d_in[7];
  const float* xp_b1    = (const float*)d_in[8];
  const float* xp_w2    = (const float*)d_in[9];
  const float* xp_b2    = (const float*)d_in[10];
  const float* ep_w     = (const float*)d_in[11];
  const float* ep_b     = (const float*)d_in[12];
  const float* vp_w     = (const float*)d_in[13];
  const float* xv_w1    = (const float*)d_in[14];
  const float* xv_b1    = (const float*)d_in[15];
  const float* xv_w2    = (const float*)d_in[16];
  const float* xv_b2    = (const float*)d_in[17];
  const float* ap_w1    = (const float*)d_in[18];
  const float* ap_b1    = (const float*)d_in[19];
  const float* ap_w2    = (const float*)d_in[20];
  const float* ap_b2    = (const float*)d_in[21];
  const float* pd_w     = (const float*)d_in[22];
  const int*   eidx     = (const int*)d_in[23];
  const int*   neigh    = (const int*)d_in[24];
  const int*   z        = (const int*)d_in[25];
  const int*   tg       = (const int*)d_in[26];
  float* out = (float*)d_out;

  size_t off = 0;
  auto alloc = [&](size_t bytes) -> void* {
    off = (off + 255) & ~(size_t)255;
    void* p = (char*)d_ws + off;
    off += bytes;
    return p;
  };
  // row-padded activation buffers (NP rows) so BM=32 GEMM needs no guards
  float* xbuf   = (float*)alloc((size_t)NP * Hh * 4);
  float* vec    = (float*)alloc((size_t)NP * H3 * 4);   // [3*NP, Hh] rows
  float* posc   = (float*)alloc((size_t)Nn * 3 * 4);
  float* shift  = (float*)alloc((size_t)Ee * 3 * 4);
  float* mid    = (float*)alloc((size_t)NP * Hhalf * 4);
  float* big768 = (float*)alloc((size_t)NP * H3 * 4);
  float* dxb    = (float*)alloc((size_t)Nn * Hh * 4);
  float* dvec   = (float*)alloc((size_t)Nn * H3 * 4);
  float* v12    = (float*)alloc((size_t)3 * NP * H2 * 4);
  float* hcat   = (float*)alloc((size_t)NP * H2 * 4);
  float* tmp1   = (float*)alloc((size_t)NP * Hh * 4);
  float* tmp2   = (float*)alloc((size_t)NP * Hh * 4);

  __bf16* combT = (__bf16*)alloc((size_t)H2 * Hh * 2);
  __bf16 *xp1T[Ll], *xp2T[Ll], *epT[Ll], *vpT[Ll], *xv1T[Ll], *xv2T[Ll], *ap1T[Ll], *ap2T[Ll];
  for (int l = 0; l < Ll; ++l) {
    xp1T[l] = (__bf16*)alloc((size_t)Hh * Hhalf * 2);
    xp2T[l] = (__bf16*)alloc((size_t)Hhalf * H3 * 2);
    epT[l]  = (__bf16*)alloc((size_t)NRBF * H3 * 2);
    vpT[l]  = (__bf16*)alloc((size_t)Hh * H2 * 2);
    xv1T[l] = (__bf16*)alloc((size_t)H2 * Hh * 2);
    xv2T[l] = (__bf16*)alloc((size_t)Hh * H3 * 2);
    ap1T[l] = (__bf16*)alloc((size_t)Hh * Hh * 2);
    ap2T[l] = (__bf16*)alloc((size_t)Hh * Hh * 2);
  }

  auto convT = [&](const float* src, __bf16* dst, int K, int Nc) {
    int total = K * Nc;
    wconvT_kernel<<<(total + 255) / 256, 256, 0, stream>>>(src, dst, K, Nc);
  };
  convT(comb_w, combT, H2, Hh);
  for (int l = 0; l < Ll; ++l) {
    convT(xp_w1 + (size_t)l * Hh * Hhalf, xp1T[l], Hh, Hhalf);
    convT(xp_w2 + (size_t)l * Hhalf * H3, xp2T[l], Hhalf, H3);
    convT(ep_w  + (size_t)l * NRBF * H3,  epT[l],  NRBF, H3);
    convT(vp_w  + (size_t)l * Hh * H2,    vpT[l],  Hh, H2);
    convT(xv_w1 + (size_t)l * H2 * Hh,    xv1T[l], H2, Hh);
    convT(xv_w2 + (size_t)l * Hh * H3,    xv2T[l], Hh, H3);
    convT(ap_w1 + (size_t)l * Hh * Hh,    ap1T[l], Hh, Hh);
    convT(ap_w2 + (size_t)l * Hh * Hh,    ap2T[l], Hh, Hh);
  }

  auto gemm = [&](const float* A, const __bf16* WT, const float* bias, float* C,
                  int Mpad, int K, int Nc, int act) {
    gemm_kernel<<<dim3(Nc / 128, Mpad / 32), 256, 0, stream>>>(A, WT, bias, C, K, Nc, act);
  };

  // init
  hipMemcpyAsync(posc, pos, (size_t)Nn * 3 * 4, hipMemcpyDeviceToDevice, stream);
  hipMemsetAsync(vec, 0, (size_t)NP * H3 * 4, stream);
  shift_kernel<<<(Ee + 255) / 256, 256, 0, stream>>>(cell, cofs, neigh, shift);
  embed_cat_kernel<<<(Nn * H2 + 255) / 256, 256, 0, stream>>>(atom_emb, tags_emb, z, tg, hcat);
  gemm(hcat, combT, comb_b, xbuf, NP, H2, Hh, 0);

  for (int l = 0; l < Ll; ++l) {
    // node projection: x_h = silu(x@w1+b1)@w2+b2
    gemm(xbuf, xp1T[l], xp_b1 + (size_t)l * Hhalf, mid, NP, Hh, Hhalf, 1);
    gemm(mid, xp2T[l], xp_b2 + (size_t)l * H3, big768, NP, Hhalf, H3, 0);
    // message passing (fused RBF+WMMA+register epilogue+scatter)
    hipMemsetAsync(dxb, 0, (size_t)Nn * Hh * 4, stream);
    hipMemsetAsync(dvec, 0, (size_t)Nn * H3 * 4, stream);
    edge_kernel<<<Ee / 16, 256, 0, stream>>>(posc, shift, eidx, big768, vec, epT[l],
                                             ep_b + (size_t)l * H3, dvec, dxb);
    combine_kernel<<<((size_t)Nn * H3 + 255) / 256, 256, 0, stream>>>(xbuf, dxb, vec, dvec);
    // self-updating
    gemm(vec, vpT[l], nullptr, v12, 3 * NP, Hh, H2, 0);
    vnorm_cat_kernel<<<((size_t)Nn * Hh + 255) / 256, 256, 0, stream>>>(xbuf, v12, hcat);
    gemm(hcat, xv1T[l], xv_b1 + (size_t)l * Hh, tmp1, NP, H2, Hh, 1);
    gemm(tmp1, xv2T[l], xv_b2 + (size_t)l * H3, big768, NP, Hh, H3, 0);
    selfupd_kernel<<<((size_t)Nn * Hh + 255) / 256, 256, 0, stream>>>(xbuf, vec, big768, v12);
    // structure updating
    gemm(xbuf, ap1T[l], ap_b1 + (size_t)l * Hh, tmp1, NP, Hh, Hh, 1);
    gemm(tmp1, ap2T[l], ap_b2 + (size_t)l * Hh, tmp2, NP, Hh, Hh, 0);
    struct_kernel<<<Nn, 256, 0, stream>>>(tmp2, vec, pd_w + (size_t)l * Hh, posc,
                                          out + (size_t)l * Nn * 3);
  }
}
